// BidirectionalLSTM_53953379172848
// MI455X (gfx1250) — matrix operand order repaired and checked by reading, not compiled, not run on hardware
//
#include <hip/hip_runtime.h>
#include <hip/hip_bf16.h>

// ---------------------------------------------------------------------------
// Bidirectional 2-layer LSTM for MI455X (gfx1250, wave32, WMMA).
//   Phase A: xg = x @ Wih^T + b  as bf16 WMMA GEMMs (M=65536, N=1024, K=256/512)
//   Phase B: persistent scan kernel, 4 WGs per direction, h double-buffered in
//            global bf16, per-step agent-scope counter barrier, c in registers.
// Per-wave tile is 32x64 (2 M-tiles x 4 N-tiles): 8 WMMA per 12 b128 loads,
// with every fragment in its own register set so loads clause + overlap WMMA.
// ---------------------------------------------------------------------------

typedef __attribute__((ext_vector_type(16))) __bf16 v16bf;
typedef __attribute__((ext_vector_type(8)))  float  v8f;

#define B_  64
#define T_  1024
#define D_  256
#define H_  256
#define G4_ 1024   // 4*H
#define NWG_DIR 4  // workgroups per direction in scan

union FragU { v16bf v; uint4 q[2]; };

// A fragment (16x32 bf16, row-major, leading dim lda).
// Lane<16:  M=lane,    K = k0+[0..7]  and k0+[16..23]
// Lane>=16: M=lane-16, K = k0+[8..15] and k0+[24..31]
__device__ __forceinline__ v16bf load_frag_a(const __bf16* A, int lda, int m0, int k0, int lane) {
  const int half = lane >> 4;
  const int r = m0 + (lane & 15);
  const __bf16* p = A + (size_t)r * lda + k0 + half * 8;
  FragU f;
  f.q[0] = *(const uint4*)(p);
  f.q[1] = *(const uint4*)(p + 16);
  return f.v;
}

// B fragment (32x16 bf16). Column n of B == row n of weight matrix W[N,K]
// (row-major, leading dim ldw), i.e. B[k][n] = W[n][k].
// Lane<16:  N=n0+lane,    K = k0+[0..7]  and k0+[16..23]
// Lane>=16: N=n0+lane-16, K = k0+[8..15] and k0+[24..31]
__device__ __forceinline__ v16bf load_frag_b(const __bf16* W, int ldw, int n0, int k0, int lane) {
  const int half = lane >> 4;
  const int n = n0 + (lane & 15);
  const __bf16* p = W + (size_t)n * ldw + k0 + half * 8;
  FragU f;
  f.q[0] = *(const uint4*)(p);
  f.q[1] = *(const uint4*)(p + 16);
  return f.v;
}

__device__ __forceinline__ v8f wmma_bf16(v16bf a, v16bf b, v8f c) {
  // v_wmma_f32_16x16x32_bf16  (neg_a, A, neg_b, B, c_mod, C, reuse_a, reuse_b)
  return __builtin_amdgcn_wmma_f32_16x16x32_bf16(false, a, false, b, (short)0, c, false, false);
}

// ---------------------------------------------------------------------------
__global__ void cvt_f32_bf16(const float* __restrict__ src, __bf16* __restrict__ dst, int n) {
  int i = 4 * (blockIdx.x * blockDim.x + threadIdx.x);
  if (i >= n) return;
  float4 v = *(const float4*)(src + i);
  __bf16 o[4] = { (__bf16)v.x, (__bf16)v.y, (__bf16)v.z, (__bf16)v.w };
  *(uint2*)(dst + i) = *(const uint2*)o;
}

// ---------------------------------------------------------------------------
// C[M,N] (fp32) = A[M,K](bf16) @ W[N,K]^T (bf16) + bias[N]
// Block = 8 waves, block tile 128x128. Wave tile 32x64:
//   wave&3 -> M sub-pair (2 x 16 rows), wave>>2 -> N strip (4 x 16 cols).
__global__ void gemm_x_wT_bias(const __bf16* __restrict__ A, const __bf16* __restrict__ W,
                               const float* __restrict__ bias, float* __restrict__ C,
                               int M, int N, int K) {
  const int lane = threadIdx.x & 31;
  const int wave = threadIdx.x >> 5;
  const int m0 = blockIdx.x * 128 + (wave & 3) * 32;   // two M tiles: m0, m0+16
  const int n0 = blockIdx.y * 128 + (wave >> 2) * 64;  // four N tiles
  if (m0 >= M) return;

  const v8f z = {0.f,0.f,0.f,0.f,0.f,0.f,0.f,0.f};
  v8f acc[2][4] = { { z, z, z, z }, { z, z, z, z } };

  for (int k0 = 0; k0 < K; k0 += 32) {
    v16bf a0 = load_frag_a(A, K, m0,      k0, lane);
    v16bf a1 = load_frag_a(A, K, m0 + 16, k0, lane);
    v16bf b0 = load_frag_b(W, K, n0,      k0, lane);
    v16bf b1 = load_frag_b(W, K, n0 + 16, k0, lane);
    v16bf b2 = load_frag_b(W, K, n0 + 32, k0, lane);
    v16bf b3 = load_frag_b(W, K, n0 + 48, k0, lane);
    acc[0][0] = wmma_bf16(a0, b0, acc[0][0]);
    acc[0][1] = wmma_bf16(a0, b1, acc[0][1]);
    acc[0][2] = wmma_bf16(a0, b2, acc[0][2]);
    acc[0][3] = wmma_bf16(a0, b3, acc[0][3]);
    acc[1][0] = wmma_bf16(a1, b0, acc[1][0]);
    acc[1][1] = wmma_bf16(a1, b1, acc[1][1]);
    acc[1][2] = wmma_bf16(a1, b2, acc[1][2]);
    acc[1][3] = wmma_bf16(a1, b3, acc[1][3]);
  }

  const int half = lane >> 4, nl = lane & 15;
#pragma unroll
  for (int u = 0; u < 2; ++u) {
    const int mb = m0 + u * 16 + half * 8;
#pragma unroll
    for (int j = 0; j < 4; ++j) {
      const int n = n0 + j * 16 + nl;
      const float bv = bias[n];
      float* cp = C + (size_t)mb * N + n;
#pragma unroll
      for (int r = 0; r < 8; ++r) cp[(size_t)r * N] = acc[u][j][r] + bv;
    }
  }
}

// ---------------------------------------------------------------------------
// Per-direction grid barrier over NWG_DIR blocks; monotonic counter.
__device__ __forceinline__ void grid_bar(unsigned int* ctr, unsigned int phase) {
  __syncthreads();
  if (threadIdx.x == 0) {
    __hip_atomic_fetch_add(ctr, 1u, __ATOMIC_RELEASE, __HIP_MEMORY_SCOPE_AGENT);
    while (__hip_atomic_load(ctr, __ATOMIC_ACQUIRE, __HIP_MEMORY_SCOPE_AGENT) <
           (unsigned)NWG_DIR * phase) {
      __builtin_amdgcn_s_sleep(1);
    }
  }
  __syncthreads();
}

// ---------------------------------------------------------------------------
// Persistent recurrent scan for one layer (both directions concurrently).
// grid = 8 blocks: blockIdx.x>>2 = dir, &3 = j-slice owner (64 hidden units).
// Wave (8 per WG) owns (M-pair, 16-col j column): wave&1 -> rows 0-31/32-63,
// (wave>>1)&3 -> 16-wide j column. It computes all 4 gate tiles for its j's,
// so gate combination + c state live entirely in registers.
__global__ void lstm_scan(const float* __restrict__ xg_f, const float* __restrict__ xg_b,
                          const __bf16* __restrict__ Whh_f, const __bf16* __restrict__ Whh_b,
                          const float* __restrict__ h0, const float* __restrict__ c0,
                          __bf16* __restrict__ hbuf,       // [2 dirs][2 phases][B_*H_] bf16
                          __bf16* __restrict__ out_bf16,   // layer0: [B,T,2H] bf16
                          float*  __restrict__ out_f32,    // layer1: [B,T,2H] fp32
                          float* __restrict__ hT, float* __restrict__ cT,  // [B,2H]
                          unsigned int* __restrict__ bar, int layer) {
  const int dir  = blockIdx.x >> 2;
  const int wgid = blockIdx.x & 3;
  const int lane = threadIdx.x & 31;
  const int wave = threadIdx.x >> 5;
  const int half = lane >> 4, nl = lane & 15;
  const int m0a   = (wave & 1) * 32;                       // M tiles m0a, m0a+16
  const int jbase = wgid * 64 + ((wave >> 1) & 3) * 16;    // hidden col base
  const int j     = jbase + nl;                            // this lane's hidden unit

  const __bf16* Whh = dir ? Whh_b : Whh_f;
  const float*  xg  = dir ? xg_b  : xg_f;
  __bf16* hb0 = hbuf + (size_t)dir * 2 * B_ * H_;
  __bf16* hb1 = hb0 + B_ * H_;
  unsigned int* ctr = bar + layer * 2 + dir;
  const int cell = dir * 2 + layer;                        // h_0/c_0 row (f0,f1,b0,b1)
  const float* h0c = h0 + (size_t)cell * B_ * H_;
  const float* c0c = c0 + (size_t)cell * B_ * H_;

  float c_reg[2][8];
  float h_last[2][8];
#pragma unroll
  for (int u = 0; u < 2; ++u) {
    const int mb = m0a + u * 16 + half * 8;
#pragma unroll
    for (int r = 0; r < 8; ++r) {
      const int b = mb + r;
      c_reg[u][r] = c0c[(size_t)b * H_ + j];
      hb0[(size_t)b * H_ + j] = (__bf16)h0c[(size_t)b * H_ + j];
      h_last[u][r] = 0.f;
    }
  }
  grid_bar(ctr, 1);   // h(0) published

  for (int s = 0; s < T_; ++s) {
    const int t = dir ? (T_ - 1 - s) : s;
    const __bf16* hin  = (s & 1) ? hb1 : hb0;
    __bf16*       hout = (s & 1) ? hb0 : hb1;

    // Prefetch next step's xg rows (global_prefetch_b8).
    if (s + 1 < T_) {
      const int tn = dir ? (T_ - 2 - s) : (s + 1);
      const float* pf = xg + ((size_t)(m0a + half * 8) * T_ + tn) * G4_ + j;
      __builtin_prefetch(pf,          0, 1);
      __builtin_prefetch(pf + H_,     0, 1);
      __builtin_prefetch(pf + 2 * H_, 0, 1);
      __builtin_prefetch(pf + 3 * H_, 0, 1);
    }

    const v8f z = {0.f,0.f,0.f,0.f,0.f,0.f,0.f,0.f};
    v8f acc[2][4] = { { z, z, z, z }, { z, z, z, z } };   // [Mtile][gate i,f,g,o]
#pragma unroll
    for (int k0 = 0; k0 < H_; k0 += 32) {
      v16bf a0 = load_frag_a(hin, H_, m0a,      k0, lane);
      v16bf a1 = load_frag_a(hin, H_, m0a + 16, k0, lane);
      v16bf g0 = load_frag_b(Whh, H_, 0 * H_ + jbase, k0, lane);
      v16bf g1 = load_frag_b(Whh, H_, 1 * H_ + jbase, k0, lane);
      v16bf g2 = load_frag_b(Whh, H_, 2 * H_ + jbase, k0, lane);
      v16bf g3 = load_frag_b(Whh, H_, 3 * H_ + jbase, k0, lane);
      acc[0][0] = wmma_bf16(a0, g0, acc[0][0]);
      acc[0][1] = wmma_bf16(a0, g1, acc[0][1]);
      acc[0][2] = wmma_bf16(a0, g2, acc[0][2]);
      acc[0][3] = wmma_bf16(a0, g3, acc[0][3]);
      acc[1][0] = wmma_bf16(a1, g0, acc[1][0]);
      acc[1][1] = wmma_bf16(a1, g1, acc[1][1]);
      acc[1][2] = wmma_bf16(a1, g2, acc[1][2]);
      acc[1][3] = wmma_bf16(a1, g3, acc[1][3]);
    }

#pragma unroll
    for (int u = 0; u < 2; ++u) {
      const int mb = m0a + u * 16 + half * 8;
#pragma unroll
      for (int r = 0; r < 8; ++r) {
        const int b = mb + r;
        const float* xr = xg + ((size_t)b * T_ + t) * G4_ + j;
        float gi = acc[u][0][r] + xr[0];
        float gf = acc[u][1][r] + xr[H_];
        float gg = acc[u][2][r] + xr[2 * H_];
        float go = acc[u][3][r] + xr[3 * H_];
        gi = 1.f / (1.f + __expf(-gi));
        gf = 1.f / (1.f + __expf(-gf));
        gg = tanhf(gg);
        go = 1.f / (1.f + __expf(-go));
        const float c = gf * c_reg[u][r] + gi * gg;
        c_reg[u][r] = c;
        const float h = go * tanhf(c);
        h_last[u][r] = h;
        hout[(size_t)b * H_ + j] = (__bf16)h;
        const size_t op = ((size_t)b * T_ + t) * (2 * H_) + dir * H_ + j;
        if (out_f32) out_f32[op] = h;
        else         out_bf16[op] = (__bf16)h;
      }
    }
    grid_bar(ctr, (unsigned)s + 2);   // h(s+1) published
  }

#pragma unroll
  for (int u = 0; u < 2; ++u) {
    const int mb = m0a + u * 16 + half * 8;
#pragma unroll
    for (int r = 0; r < 8; ++r) {
      const int b = mb + r;
      hT[(size_t)b * 2 * H_ + dir * H_ + j] = h_last[u][r];
      cT[(size_t)b * 2 * H_ + dir * H_ + j] = c_reg[u][r];
    }
  }
}

// ---------------------------------------------------------------------------
extern "C" void kernel_launch(void* const* d_in, const int* in_sizes, int n_in,
                              void* d_out, int out_size, void* d_ws, size_t ws_size,
                              hipStream_t stream) {
  (void)in_sizes; (void)n_in; (void)out_size; (void)ws_size;
  const float* x      = (const float*)d_in[0];
  const float* h0     = (const float*)d_in[1];
  const float* c0     = (const float*)d_in[2];
  const float* Wih_f0 = (const float*)d_in[3];
  const float* Whh_f0 = (const float*)d_in[4];
  const float* b_f0   = (const float*)d_in[5];
  const float* Wih_b0 = (const float*)d_in[6];
  const float* Whh_b0 = (const float*)d_in[7];
  const float* b_b0   = (const float*)d_in[8];
  const float* Wih_f1 = (const float*)d_in[9];
  const float* Whh_f1 = (const float*)d_in[10];
  const float* b_f1   = (const float*)d_in[11];
  const float* Wih_b1 = (const float*)d_in[12];
  const float* Whh_b1 = (const float*)d_in[13];
  const float* b_b1   = (const float*)d_in[14];

  char* ws = (char*)d_ws;
  size_t off = 0;
  auto alloc = [&](size_t bytes) -> char* {
    char* p = ws + off;
    off = (off + bytes + 255) & ~(size_t)255;
    return p;
  };
  const size_t M = (size_t)B_ * T_;  // 65536
  float*  xg_f  = (float*) alloc(M * G4_ * sizeof(float));   // 256 MB
  float*  xg_b  = (float*) alloc(M * G4_ * sizeof(float));   // 256 MB
  __bf16* seq0  = (__bf16*)alloc(M * D_ * 2);                //  32 MB
  __bf16* seq1  = (__bf16*)alloc(M * (2 * H_) * 2);          //  64 MB
  __bf16* wihf0 = (__bf16*)alloc((size_t)G4_ * D_ * 2);
  __bf16* wihb0 = (__bf16*)alloc((size_t)G4_ * D_ * 2);
  __bf16* whhf0 = (__bf16*)alloc((size_t)G4_ * H_ * 2);
  __bf16* whhb0 = (__bf16*)alloc((size_t)G4_ * H_ * 2);
  __bf16* wihf1 = (__bf16*)alloc((size_t)G4_ * 2 * H_ * 2);
  __bf16* wihb1 = (__bf16*)alloc((size_t)G4_ * 2 * H_ * 2);
  __bf16* whhf1 = (__bf16*)alloc((size_t)G4_ * H_ * 2);
  __bf16* whhb1 = (__bf16*)alloc((size_t)G4_ * H_ * 2);
  __bf16* hbuf  = (__bf16*)alloc((size_t)2 * 2 * B_ * H_ * 2);
  unsigned int* bar = (unsigned int*)alloc(256);

  hipMemsetAsync(bar, 0, 256, stream);   // capture-safe counter reset

  auto cvt = [&](const float* s, __bf16* d, int n) {
    cvt_f32_bf16<<<(n / 4 + 255) / 256, 256, 0, stream>>>(s, d, n);
  };
  cvt(x,      seq0,  (int)(M * D_));
  cvt(Wih_f0, wihf0, G4_ * D_);
  cvt(Wih_b0, wihb0, G4_ * D_);
  cvt(Whh_f0, whhf0, G4_ * H_);
  cvt(Whh_b0, whhb0, G4_ * H_);
  cvt(Wih_f1, wihf1, G4_ * 2 * H_);
  cvt(Wih_b1, wihb1, G4_ * 2 * H_);
  cvt(Whh_f1, whhf1, G4_ * H_);
  cvt(Whh_b1, whhb1, G4_ * H_);

  float* out   = (float*)d_out;
  const size_t seqN = M * 2 * H_;                    // [B,T,2H]
  float* hcopy = out + seqN;                         // [2,B,2H]
  float* ccopy = hcopy + (size_t)2 * B_ * 2 * H_;    // [2,B,2H]

  const dim3 g1((unsigned)(M / 128), G4_ / 128);     // (512, 8)

  // ---- Layer 0 ----
  gemm_x_wT_bias<<<g1, 256, 0, stream>>>(seq0, wihf0, b_f0, xg_f, (int)M, G4_, D_);
  gemm_x_wT_bias<<<g1, 256, 0, stream>>>(seq0, wihb0, b_b0, xg_b, (int)M, G4_, D_);
  lstm_scan<<<2 * NWG_DIR, 256, 0, stream>>>(xg_f, xg_b, whhf0, whhb0, h0, c0, hbuf,
                                             seq1, nullptr,
                                             hcopy, ccopy, bar, 0);
  // ---- Layer 1 ---- (xg buffers reused)
  gemm_x_wT_bias<<<g1, 256, 0, stream>>>(seq1, wihf1, b_f1, xg_f, (int)M, G4_, 2 * H_);
  gemm_x_wT_bias<<<g1, 256, 0, stream>>>(seq1, wihb1, b_b1, xg_b, (int)M, G4_, 2 * H_);
  lstm_scan<<<2 * NWG_DIR, 256, 0, stream>>>(xg_f, xg_b, whhf1, whhb1, h0, c0, hbuf,
                                             nullptr, out,
                                             hcopy + (size_t)B_ * 2 * H_,
                                             ccopy + (size_t)B_ * 2 * H_, bar, 1);
}
